// LayerwisePathwayMLP_33981781246563
// MI455X (gfx1250) — matrix operand order repaired and checked by reading, not compiled, not run on hardware
//
#include <hip/hip_runtime.h>

typedef __bf16        v16bf __attribute__((ext_vector_type(16)));
typedef float         v8f   __attribute__((ext_vector_type(8)));
typedef unsigned int  v8u   __attribute__((ext_vector_type(8)));

__device__ __forceinline__ unsigned short f2bf(float f) {
  unsigned int u = __builtin_bit_cast(unsigned int, f);
  u += 0x7FFFu + ((u >> 16) & 1u);          // round-to-nearest-even
  return (unsigned short)(u >> 16);
}
__device__ __forceinline__ unsigned int pack2bf(float lo, float hi) {
  return (unsigned int)f2bf(lo) | ((unsigned int)f2bf(hi) << 16);
}
__device__ __forceinline__ float bflo(unsigned int u) {
  return __builtin_bit_cast(float, u << 16);
}
__device__ __forceinline__ float bfhi(unsigned int u) {
  return __builtin_bit_cast(float, u & 0xFFFF0000u);
}

struct Params {
  const float* x;
  const float* b[6];
  const float* rW[6];
  const float* rb[6];
  const unsigned short* wsl;   // prepacked bf16 weight slices in d_ws
  float* out;
};

// output-group index for column n of layer l; -1 => column gets zero gate
__device__ __forceinline__ int jmap(int l, int n, int dout) {
  if (l < 5) return (n == 0) ? -1 : ((n >= (dout >> 1)) ? 1 : 0);
  if (n == 1) return 0;
  if (n >= 2 && n <= 3) return 1;
  if (n >= 4 && n <= 5) return 2;
  if (n >= 6 && n <= 9) return 3;
  return -1;
}

// ---------------- weight repack pre-pass ----------------
// dst is doutp x kp bf16, row-major over n; k = ka + krel, masked to [gs, ge)
__global__ __launch_bounds__(256) void prep_w(
    const float* __restrict__ W, unsigned short* __restrict__ dst,
    int din, int dout, int doutp, int kp, int ka, int gs, int ge) {
  int idx = blockIdx.x * 256 + threadIdx.x;
  int tot = doutp * kp;
  if (idx >= tot) return;
  int n = idx / kp;
  int k = ka + (idx - n * kp);
  float v = 0.f;
  if (n < dout && k >= gs && k < ge) v = W[(size_t)n * din + k];
  dst[idx] = f2bf(v);
}

// ---------------- fused layer step ----------------
// gates layout: transposed [g][64] so C-tile gate fetch = 2x ds_load_b128
__device__ __forceinline__ void layer_step(
    int l,
    const unsigned short* actIn, int SIN,
    unsigned short* actOut, int SOUT,
    int din, int dout, int doutp, int ni, int no, int ng, int kp,
    int ka0, int ka1, int ka2, int ka3,
    const unsigned short* wbase,
    const float* rW, const float* rb, const float* bias,
    float* gates, float* outG, int row0, int tid)
{
  // ----- router logits (vectorized: 8 bf16 + 8 f32 per step) -----
  {
    int m = tid & 63;
    const unsigned short* arow = actIn + (size_t)m * SIN;
    for (int g = tid >> 6; g < ng; g += 4) {
      float s = rb[g];
      const float* wr = rW + (size_t)g * din;
      for (int k = 0; k < din; k += 8) {
        uint4  a  = *(const uint4*)(arow + k);
        float4 w0 = *(const float4*)(wr + k);
        float4 w1 = *(const float4*)(wr + k + 4);
        s += bflo(a.x) * w0.x + bfhi(a.x) * w0.y
           + bflo(a.y) * w0.z + bfhi(a.y) * w0.w
           + bflo(a.z) * w1.x + bfhi(a.z) * w1.y
           + bflo(a.w) * w1.z + bfhi(a.w) * w1.w;
      }
      gates[g * 64 + m] = s;
    }
  }
  __syncthreads();
  // ----- softmax per row -----
  if (tid < 64) {
    float mx = gates[tid];
    for (int g = 1; g < ng; ++g) mx = fmaxf(mx, gates[g * 64 + tid]);
    float ssum = 0.f;
    for (int g = 0; g < ng; ++g) {
      float e = expf(gates[g * 64 + tid] - mx);
      gates[g * 64 + tid] = e;
      ssum += e;
    }
    float inv = 1.f / ssum;
    for (int g = 0; g < ng; ++g) gates[g * 64 + tid] *= inv;
  }
  __syncthreads();

  // ----- gated grouped GEMM via bf16 WMMA -----
  int wave = tid >> 5;
  int lane = tid & 31;
  int hh   = lane >> 4;      // K-half selector
  int l16  = lane & 15;
  int NT    = doutp >> 4;
  int nJobs = 4 * NT;        // 4 M-tiles of 16 rows x NT N-tiles
  int ka[4] = {ka0, ka1, ka2, ka3};

  for (int job = wave; job < nJobs; job += 8) {
    int mt = job / NT, nt = job - mt * NT;
    int mbase = mt << 4;
    int ncol  = (nt << 4) + l16;
    int j = jmap(l, ncol, dout);
    float msk = (j >= 0) ? 1.f : 0.f;
    int   jc  = (j >= 0) ? j : 0;

    v8f cfin = {0.f,0.f,0.f,0.f,0.f,0.f,0.f,0.f};
    float gacc[8] = {0.f,0.f,0.f,0.f,0.f,0.f,0.f,0.f};

    const unsigned short* aRow = actIn + (size_t)(mbase + l16) * SIN;

    #pragma unroll
    for (int i = 0; i < 4; ++i) {
      if (i < ni) {
        v8f cp = {0.f,0.f,0.f,0.f,0.f,0.f,0.f,0.f};
        const unsigned short* wrow = wbase + (size_t)i * doutp * kp + (size_t)ncol * kp;
        int kbase = ka[i];
        int nk = kp >> 5;
        #pragma unroll
        for (int kk = 0; kk < 8; ++kk) {
          if (kk < nk) {
            // A fragment: lanes 0-15 M=0..15; VGPR 0-3 = K hh*8+[0..7], VGPR 4-7 = K 16+hh*8+[0..7]
            const unsigned short* ap = aRow + kbase + (kk << 5) + (hh << 3);
            uint4 a0 = *(const uint4*)(ap);
            uint4 a1 = *(const uint4*)(ap + 16);
            // B fragment: lane N=lane%16; VGPR 0-3 = K hh*8+[0..7], VGPR 4-7 = K 16+hh*8+[0..7]
            const unsigned short* bp = wrow + (kk << 5) + (hh << 3);
            uint4 b0 = *(const uint4*)(bp);
            uint4 b1 = *(const uint4*)(bp + 16);
            v8u au; au[0]=a0.x; au[1]=a0.y; au[2]=a0.z; au[3]=a0.w;
                    au[4]=a1.x; au[5]=a1.y; au[6]=a1.z; au[7]=a1.w;
            v8u bu; bu[0]=b0.x; bu[1]=b0.y; bu[2]=b0.z; bu[3]=b0.w;
                    bu[4]=b1.x; bu[5]=b1.y; bu[6]=b1.z; bu[7]=b1.w;
            v16bf A = __builtin_bit_cast(v16bf, au);
            v16bf B = __builtin_bit_cast(v16bf, bu);
            cp = __builtin_amdgcn_wmma_f32_16x16x32_bf16(
                false, A, false, B, (short)0, cp, false, false);
          }
        }
        // fold gated partial: C VGPR v holds row (mbase + v + 8*hh), col ncol.
        // gates transposed: rows for v=0..7 are 8 consecutive floats -> 2x b128.
        const float* gp = gates + (i * no + jc) * 64 + mbase + (hh << 3);
        float4 g0 = *(const float4*)(gp);
        float4 g1 = *(const float4*)(gp + 4);
        float gv[8] = {g0.x, g0.y, g0.z, g0.w, g1.x, g1.y, g1.z, g1.w};
        #pragma unroll
        for (int v = 0; v < 8; ++v) {
          float gm = gv[v] * msk;
          cfin[v] += cp[v] * gm;
          gacc[v] += gm;
        }
      }
    }

    float bn = (ncol < dout) ? bias[ncol] : 0.f;
    #pragma unroll
    for (int v = 0; v < 8; ++v) {
      float val = cfin[v] + bn * gacc[v];
      if (l < 5) val = 0.5f * val * (1.f + erff(val * 0.70710678118654752f));
      int m = mbase + v + (hh << 3);
      if (l == 5) {
        if (ncol < 10) outG[(size_t)(row0 + m) * 10 + ncol] = val;
      } else {
        actOut[m * SOUT + ncol] = f2bf(val);
      }
    }
  }
  __syncthreads();
}

// ---------------- fused 6-layer kernel: 64 batch rows / workgroup ----------------
__global__ __launch_bounds__(256) void fused_mlp(Params p) {
  extern __shared__ char smem[];
  unsigned short* bufA = (unsigned short*)smem;                        // 64 x 800 bf16
  unsigned short* bufB = (unsigned short*)(smem + 64 * 800 * 2);       // 64 x 512 bf16
  float* gates = (float*)(smem + 64 * 800 * 2 + 64 * 512 * 2);         // 8 x 64 f32 (transposed)

  int tid  = threadIdx.x;
  int row0 = blockIdx.x * 64;

  // stage x -> bufA as bf16 (vectorized), zero-pad cols 784..799
  for (int idx = tid; idx < 64 * 200; idx += 256) {
    int m = idx / 200, k4 = (idx - m * 200) << 2;
    uint2 pk;
    if (k4 < 784) {
      float4 v = *(const float4*)(p.x + (size_t)(row0 + m) * 784 + k4);
      pk.x = pack2bf(v.x, v.y);
      pk.y = pack2bf(v.z, v.w);
    } else {
      pk.x = 0u; pk.y = 0u;
    }
    *(uint2*)(bufA + m * 800 + k4) = pk;
  }
  __syncthreads();

  const unsigned short* ws = p.wsl;
  layer_step(0, bufA, 800, bufB, 512, 784, 512, 512, 4, 2, 8, 224, 0,192,384,576,
             ws + 0,      p.rW[0], p.rb[0], p.b[0], gates, nullptr, row0, tid);
  layer_step(1, bufB, 512, bufA, 800, 512, 256, 256, 2, 2, 4, 256, 0,256,0,0,
             ws + 458752, p.rW[1], p.rb[1], p.b[1], gates, nullptr, row0, tid);
  layer_step(2, bufA, 800, bufB, 512, 256, 128, 128, 2, 2, 4, 128, 0,128,0,0,
             ws + 589824, p.rW[2], p.rb[2], p.b[2], gates, nullptr, row0, tid);
  layer_step(3, bufB, 512, bufA, 800, 128,  64,  64, 2, 2, 4,  64, 0,64,0,0,
             ws + 622592, p.rW[3], p.rb[3], p.b[3], gates, nullptr, row0, tid);
  layer_step(4, bufA, 800, bufB, 512,  64,  32,  32, 2, 2, 4,  32, 0,32,0,0,
             ws + 630784, p.rW[4], p.rb[4], p.b[4], gates, nullptr, row0, tid);
  layer_step(5, bufB, 512, nullptr, 0,  32,  10,  16, 2, 4, 8,  32, 0,0,0,0,
             ws + 632832, p.rW[5], p.rb[5], p.b[5], gates, p.out, row0, tid);
}

extern "C" void kernel_launch(void* const* d_in, const int* in_sizes, int n_in,
                              void* d_out, int out_size, void* d_ws, size_t ws_size,
                              hipStream_t stream) {
  Params P;
  P.x = (const float*)d_in[0];
  for (int l = 0; l < 6; ++l) {
    P.b[l]  = (const float*)d_in[7 + l];
    P.rW[l] = (const float*)d_in[13 + l];
    P.rb[l] = (const float*)d_in[19 + l];
  }
  unsigned short* ws = (unsigned short*)d_ws;
  P.wsl = ws;
  P.out = (float*)d_out;

  static const int DIN_[6]  = {784,512,256,128,64,32};
  static const int DOUT_[6] = {512,256,128,64,32,10};
  static const int DOUTP_[6]= {512,256,128,64,32,16};
  static const int NIv[6]   = {4,2,2,2,2,2};
  static const int KPv[6]   = {224,256,128,64,32,32};
  static const int KAv[6][4]= {{0,192,384,576},{0,256,0,0},{0,128,0,0},{0,64,0,0},{0,32,0,0},{0,0,0,0}};
  static const int GSv[6][4]= {{1,196,392,588},{1,256,0,0},{1,128,0,0},{1,64,0,0},{1,32,0,0},{1,16,0,0}};
  static const int GEv[6][4]= {{196,392,588,784},{256,512,0,0},{128,256,0,0},{64,128,0,0},{32,64,0,0},{16,32,0,0}};
  static const long OFFv[6] = {0,458752,589824,622592,630784,632832};

  for (int l = 0; l < 6; ++l) {
    const float* W = (const float*)d_in[1 + l];
    for (int i = 0; i < NIv[l]; ++i) {
      unsigned short* dst = ws + OFFv[l] + (long)i * DOUTP_[l] * KPv[l];
      int tot = DOUTP_[l] * KPv[l];
      prep_w<<<(tot + 255) / 256, 256, 0, stream>>>(
          W, dst, DIN_[l], DOUT_[l], DOUTP_[l], KPv[l],
          KAv[l][i], GSv[l][i], GEv[l][i]);
    }
  }

  size_t smemBytes = 64 * 800 * 2 + 64 * 512 * 2 + 8 * 64 * 4; // 169984 B
  fused_mlp<<<65536 / 64, 256, smemBytes, stream>>>(P);
}
